// GPT_19378892439957
// MI455X (gfx1250) — matrix operand order repaired and checked by reading, not compiled, not run on hardware
//
#include <hip/hip_runtime.h>

// ---------------- model constants ----------------
#define Lc   12
#define Dc   1024
#define Hc   16
#define Vc   50257
#define Tc   256
#define Bc   2
#define Sc   768          // 3*T
#define HDc  64

// ---------------- vector types for WMMA ----------------
typedef __bf16 bf16;
typedef bf16  v16bf __attribute__((ext_vector_type(16)));
typedef bf16  v8bf  __attribute__((ext_vector_type(8)));
typedef bf16  v4bf  __attribute__((ext_vector_type(4)));
typedef float v8f   __attribute__((ext_vector_type(8)));

union V16U { v16bf v; v8bf h[2]; };

__device__ __forceinline__ bf16 f2bf(float f) { return (bf16)f; }

__device__ __forceinline__ float gelu_f(float v) {
  float c = v + 0.044715f * v * v * v;
  return 0.5f * v * (1.0f + tanhf(0.7978845608028654f * c));
}

// ---------------- generic WMMA GEMM ----------------
// C[z][m][n] = epilogue(alpha * sum_k A[z][m][k] * B[z][k][n])
// TRANSB==true : B is stored [N][K] row-major (ldb = row stride in K units)
// EPI bit0: +bias[n], bit1: += C (residual), bit2: gelu (applied after bias)
// Block tile 128(M) x 256(N) x 32(K); 8 waves in 2x4; wave tile 64x64 = 16 WMMA.
#define BMt 128
#define BNt 256
#define BKt 32
#define BKP 40   // padded LDS row stride in bf16 elems (80 bytes, 16B aligned)

template<int EPI, bool TRANSB>
__global__ __launch_bounds__(256) void gemm_wmma(
    const float* __restrict__ A, const float* __restrict__ Bm,
    float* __restrict__ C, const float* __restrict__ bias,
    int M, int N, int K, int lda, int ldb, int ldc,
    long long sA0, long long sA1, long long sB0, long long sB1,
    long long sC0, long long sC1, int zdiv, float alpha)
{
  __shared__ __align__(16) bf16 lA[BMt][BKP];
  __shared__ __align__(16) bf16 lB[BNt][BKP];

  const int z  = blockIdx.z;
  const int z0 = z / zdiv, z1 = z % zdiv;
  A  += (long long)z0 * sA0 + (long long)z1 * sA1;
  Bm += (long long)z0 * sB0 + (long long)z1 * sB1;
  C  += (long long)z0 * sC0 + (long long)z1 * sC1;

  const int tid   = threadIdx.x;
  const int lane  = tid & 31;
  const int wave  = tid >> 5;       // 0..7
  const int waveM = wave >> 2;      // 0..1 -> 64 rows each
  const int waveN = wave & 3;       // 0..3 -> 64 cols each
  const int half  = lane >> 4;
  const int l15   = lane & 15;

  const int mBase = blockIdx.y * BMt;
  const int nBase = blockIdx.x * BNt;

  v8f acc[4][4];
#pragma unroll
  for (int mt = 0; mt < 4; ++mt)
#pragma unroll
    for (int nt = 0; nt < 4; ++nt)
      acc[mt][nt] = (v8f){0.f,0.f,0.f,0.f,0.f,0.f,0.f,0.f};

  for (int k0 = 0; k0 < K; k0 += BKt) {
    // ---- prefetch next A tile (global_prefetch_b8 path) ----
    if (k0 + BKt < K) {
      int prow = tid >> 1;
      if (mBase + prow < M)
        __builtin_prefetch(A + (size_t)(mBase + prow) * lda + k0 + BKt, 0, 1);
    }

    // ---- load A tile 128x32 fp32 -> bf16 LDS (packed b64 stores) ----
#pragma unroll
    for (int i = 0; i < 4; ++i) {
      int idx4 = i * 256 + tid;         // 1024 float4 = 4096 floats
      int row  = idx4 >> 3;             // /8 (8 float4 per 32-float row)
      int kc   = (idx4 & 7) << 2;       // *4
      float4 f = make_float4(0.f, 0.f, 0.f, 0.f);
      int gm = mBase + row;
      if (gm < M) f = *(const float4*)(A + (size_t)gm * lda + k0 + kc);
      v4bf t;
      t.x = f2bf(f.x); t.y = f2bf(f.y); t.z = f2bf(f.z); t.w = f2bf(f.w);
      *(v4bf*)&lA[row][kc] = t;
    }

    // ---- load B tile -> bf16 LDS stored [n][k] (packed b64 stores) ----
    if (TRANSB) {
#pragma unroll
      for (int i = 0; i < 8; ++i) {
        int idx4 = i * 256 + tid;       // 2048 float4 = 8192 floats (256x32)
        int row  = idx4 >> 3;           // n within tile
        int kc   = (idx4 & 7) << 2;
        float4 f = make_float4(0.f, 0.f, 0.f, 0.f);
        int gn = nBase + row;
        if (gn < N) f = *(const float4*)(Bm + (size_t)gn * ldb + k0 + kc);
        v4bf t;
        t.x = f2bf(f.x); t.y = f2bf(f.y); t.z = f2bf(f.z); t.w = f2bf(f.w);
        *(v4bf*)&lB[row][kc] = t;
      }
    } else {
      // B global [K,N]: each thread owns column n=tid, loads 4 coalesced rows
      // per iter and writes one transposed packed b64.
      int gn = nBase + tid;             // tid in [0,256) == column within tile
      bool ok = gn < N;
#pragma unroll
      for (int i = 0; i < 8; ++i) {
        int kb = i * 4;
        float f0 = 0.f, f1 = 0.f, f2 = 0.f, f3 = 0.f;
        if (ok) {
          const float* bp = Bm + (size_t)(k0 + kb) * ldb + gn;
          f0 = bp[0];
          f1 = bp[(size_t)ldb];
          f2 = bp[(size_t)2 * ldb];
          f3 = bp[(size_t)3 * ldb];
        }
        v4bf t;
        t.x = f2bf(f0); t.y = f2bf(f1); t.z = f2bf(f2); t.w = f2bf(f3);
        *(v4bf*)&lB[tid][kb] = t;
      }
    }
    __syncthreads();

    // ---- build fragments per ISA 16-bit A/B layouts ----
    v16bf afr[4], bfr[4];
#pragma unroll
    for (int mt = 0; mt < 4; ++mt) {
      int m = waveM * 64 + mt * 16 + l15;
      V16U u;
      u.h[0] = *(const v8bf*)&lA[m][half * 8];
      u.h[1] = *(const v8bf*)&lA[m][16 + half * 8];
      afr[mt] = u.v;
    }
#pragma unroll
    for (int nt = 0; nt < 4; ++nt) {
      int n = waveN * 64 + nt * 16 + l15;
      V16U u;
      u.h[0] = *(const v8bf*)&lB[n][half * 8];
      u.h[1] = *(const v8bf*)&lB[n][16 + half * 8];
      bfr[nt] = u.v;
    }

#pragma unroll
    for (int mt = 0; mt < 4; ++mt)
#pragma unroll
      for (int nt = 0; nt < 4; ++nt)
        acc[mt][nt] = __builtin_amdgcn_wmma_f32_16x16x32_bf16(
            false, afr[mt], false, bfr[nt], (short)0, acc[mt][nt], false, false);

    __syncthreads();
  }

  // ---- epilogue: C/D layout lane l -> col=l&15, rows r+8*(l>>4) ----
#pragma unroll
  for (int mt = 0; mt < 4; ++mt) {
#pragma unroll
    for (int nt = 0; nt < 4; ++nt) {
      int col = nBase + waveN * 64 + nt * 16 + l15;
      if (col < N) {
        v8f a = acc[mt][nt];
#pragma unroll
        for (int r = 0; r < 8; ++r) {
          int m = mBase + waveM * 64 + mt * 16 + half * 8 + r;
          if (m < M) {
            float v = alpha * a[r];
            if (EPI & 1) v += bias[col];
            if (EPI & 4) v = gelu_f(v);
            size_t idx = (size_t)m * ldc + col;
            if (EPI & 2) v += C[idx];
            C[idx] = v;
          }
        }
      }
    }
  }
}

// ---------------- embedding ----------------
__global__ __launch_bounds__(256) void embed_k(
    const float* __restrict__ states, const int* __restrict__ actions,
    const float* __restrict__ rtgs,   const int* __restrict__ tsteps,
    const float* __restrict__ se_w, const float* __restrict__ se_b,
    const float* __restrict__ re_w, const float* __restrict__ re_b,
    const float* __restrict__ ae,   const float* __restrict__ pe,
    const float* __restrict__ gpe,  float* __restrict__ x)
{
  int idx = blockIdx.x * 256 + threadIdx.x;
  if (idx >= Bc * Sc * Dc) return;
  int d  = idx & (Dc - 1);
  int bs = idx >> 10;
  int s  = bs % Sc;
  int b  = bs / Sc;
  int t  = s / 3;
  int r  = s % 3;                 // 0=rtg, 1=state, 2=action (interleave order)
  float tok;
  if (r == 0)      tok = rtgs[b * Tc + t]   * re_w[d] + re_b[d];
  else if (r == 1) tok = states[b * Tc + t] * se_w[d] + se_b[d];
  else             tok = ae[(size_t)actions[b * Tc + t] * Dc + d];
  float pos = gpe[(size_t)tsteps[b] * Dc + d] + pe[(size_t)s * Dc + d];
  x[idx] = tok + pos;
}

// ---------------- layernorm (one block per row of 1024) ----------------
__global__ __launch_bounds__(256) void layernorm_k(
    const float* __restrict__ in, float* __restrict__ out,
    const float* __restrict__ w, const float* __restrict__ b)
{
  __shared__ float red[256];
  int row = blockIdx.x;
  int tid = threadIdx.x;
  const float* p = in + (size_t)row * Dc;
  float s = 0.f, s2 = 0.f;
  for (int i = tid; i < Dc; i += 256) { float v = p[i]; s += v; s2 += v * v; }
  red[tid] = s; __syncthreads();
  for (int o = 128; o > 0; o >>= 1) { if (tid < o) red[tid] += red[tid + o]; __syncthreads(); }
  float mean = red[0] * (1.f / Dc);
  __syncthreads();
  red[tid] = s2; __syncthreads();
  for (int o = 128; o > 0; o >>= 1) { if (tid < o) red[tid] += red[tid + o]; __syncthreads(); }
  float var = red[0] * (1.f / Dc) - mean * mean;
  float rstd = rsqrtf(var + 1e-5f);
  float* q = out + (size_t)row * Dc;
  for (int i = tid; i < Dc; i += 256)
    q[i] = (p[i] - mean) * rstd * w[i] + b[i];
}

// ---------------- causal softmax over att rows ----------------
__global__ __launch_bounds__(256) void softmax_causal_k(float* __restrict__ att)
{
  __shared__ float red[256];
  int row = blockIdx.x;             // b*H*S + h*S + i
  int i   = row % Sc;
  int tid = threadIdx.x;
  float* p = att + (size_t)row * Sc;
  int n = i + 1;                    // valid (unmasked) length
  float m = -1e30f;
  for (int j = tid; j < n; j += 256) m = fmaxf(m, p[j]);
  red[tid] = m; __syncthreads();
  for (int o = 128; o > 0; o >>= 1) { if (tid < o) red[tid] = fmaxf(red[tid], red[tid + o]); __syncthreads(); }
  m = red[0];
  __syncthreads();
  float s = 0.f;
  for (int j = tid; j < n; j += 256) s += __expf(p[j] - m);
  red[tid] = s; __syncthreads();
  for (int o = 128; o > 0; o >>= 1) { if (tid < o) red[tid] += red[tid + o]; __syncthreads(); }
  float inv = 1.f / red[0];
  for (int j = tid; j < Sc; j += 256)
    p[j] = (j < n) ? __expf(p[j] - m) * inv : 0.f;
}

// ---------------- gather state-position rows ----------------
__global__ __launch_bounds__(256) void gather_states_k(
    const float* __restrict__ xln, float* __restrict__ xs)
{
  int idx = blockIdx.x * 256 + threadIdx.x;
  if (idx >= Bc * Tc * Dc) return;
  int d  = idx & (Dc - 1);
  int bt = idx >> 10;
  int t  = bt % Tc;
  int b  = bt / Tc;
  xs[idx] = xln[((size_t)(b * Sc + 3 * t + 1)) * Dc + d];
}

// ---------------- host ----------------
static inline int tilesM128(int a) { return (a + 127) / 128; }
static inline int tilesN256(int a) { return (a + 255) / 256; }

extern "C" void kernel_launch(void* const* d_in, const int* in_sizes, int n_in,
                              void* d_out, int out_size, void* d_ws, size_t ws_size,
                              hipStream_t stream)
{
  const float* states  = (const float*)d_in[0];
  const int*   actions = (const int*)  d_in[1];
  const float* rtgs    = (const float*)d_in[2];
  const int*   tsteps  = (const int*)  d_in[3];
  const float* se_w = (const float*)d_in[4];
  const float* se_b = (const float*)d_in[5];
  const float* re_w = (const float*)d_in[6];
  const float* re_b = (const float*)d_in[7];
  const float* ae   = (const float*)d_in[8];
  const float* pe   = (const float*)d_in[9];
  const float* gpe  = (const float*)d_in[10];
  const float* ln1_w  = (const float*)d_in[11];
  const float* ln1_b  = (const float*)d_in[12];
  const float* qkv_w  = (const float*)d_in[13];
  const float* qkv_b  = (const float*)d_in[14];
  const float* proj_w = (const float*)d_in[15];
  const float* proj_b = (const float*)d_in[16];
  const float* ln2_w  = (const float*)d_in[17];
  const float* ln2_b  = (const float*)d_in[18];
  const float* fc_w   = (const float*)d_in[19];
  const float* fc_b   = (const float*)d_in[20];
  const float* fcp_w  = (const float*)d_in[21];
  const float* fcp_b  = (const float*)d_in[22];
  const float* lnf_w  = (const float*)d_in[23];
  const float* lnf_b  = (const float*)d_in[24];
  const float* lm_w   = (const float*)d_in[25];
  float* out = (float*)d_out;

  const int BS  = Bc * Sc;            // 1536 rows for fused-batch GEMMs
  const int D3  = 3 * Dc;             // 3072
  const int D4  = 4 * Dc;             // 4096

  // workspace layout (fp32 elements)
  float* p    = (float*)d_ws;
  float* x    = p; p += (size_t)BS * Dc;          // residual stream
  float* xln  = p; p += (size_t)BS * Dc;          // layernorm output
  float* qkvb = p; p += (size_t)BS * D3;          // q|k|v
  float* attb = p; p += (size_t)Bc * Hc * Sc * Sc;// attention matrices
  float* yb   = p; p += (size_t)BS * Dc;          // attention output
  float* h4   = p; p += (size_t)BS * D4;          // MLP hidden
  float* xs   = p; p += (size_t)Bc * Tc * Dc;     // gathered state rows

  dim3 blk(256);

  // ---- embeddings ----
  {
    int tot = Bc * Sc * Dc;
    embed_k<<<dim3((tot + 255) / 256), blk, 0, stream>>>(
        states, actions, rtgs, tsteps, se_w, se_b, re_w, re_b, ae, pe, gpe, x);
  }

  const long long sAb = (long long)Sc * D3;          // per-batch stride into qkv
  const long long sAtt = (long long)Sc * Sc;         // per-(b,h) att stride
  const float scale = 0.125f;                        // 1/sqrt(64)

  for (int l = 0; l < Lc; ++l) {
    const float* Wq = qkv_w  + (size_t)l * Dc * D3;
    const float* bq = qkv_b  + (size_t)l * D3;
    const float* Wp = proj_w + (size_t)l * Dc * Dc;
    const float* bp = proj_b + (size_t)l * Dc;
    const float* Wf = fc_w   + (size_t)l * Dc * D4;
    const float* bf = fc_b   + (size_t)l * D4;
    const float* Wg = fcp_w  + (size_t)l * D4 * Dc;
    const float* bg = fcp_b  + (size_t)l * Dc;

    // LN1
    layernorm_k<<<dim3(BS), blk, 0, stream>>>(x, xln, ln1_w + l * Dc, ln1_b + l * Dc);

    // QKV: [1536,1024] x [1024,3072] + bias
    gemm_wmma<1, false><<<dim3(tilesN256(D3), tilesM128(BS), 1), blk, 0, stream>>>(
        xln, Wq, qkvb, bq, BS, D3, Dc, Dc, D3, D3,
        0, 0, 0, 0, 0, 0, 1, 1.0f);

    // scores: per (b,h): Q[768,64] x K^T -> att[768,768], alpha = 1/sqrt(hd)
    gemm_wmma<0, true><<<dim3(tilesN256(Sc), tilesM128(Sc), Bc * Hc), blk, 0, stream>>>(
        qkvb, qkvb + Dc, attb, nullptr, Sc, Sc, HDc, D3, D3, Sc,
        sAb, HDc, sAb, HDc, (long long)Hc * Sc * Sc, sAtt, Hc, scale);

    // causal softmax
    softmax_causal_k<<<dim3(Bc * Hc * Sc), blk, 0, stream>>>(attb);

    // att @ V: per (b,h): [768,768] x [768,64] -> y[b, s, h*64 + n]
    gemm_wmma<0, false><<<dim3(tilesN256(HDc), tilesM128(Sc), Bc * Hc), blk, 0, stream>>>(
        attb, qkvb + 2 * Dc, yb, nullptr, Sc, HDc, Sc, Sc, D3, Dc,
        (long long)Hc * Sc * Sc, sAtt, sAb, HDc, (long long)Sc * Dc, HDc, Hc, 1.0f);

    // proj + bias + residual into x
    gemm_wmma<3, false><<<dim3(tilesN256(Dc), tilesM128(BS), 1), blk, 0, stream>>>(
        yb, Wp, x, bp, BS, Dc, Dc, Dc, Dc, Dc,
        0, 0, 0, 0, 0, 0, 1, 1.0f);

    // LN2
    layernorm_k<<<dim3(BS), blk, 0, stream>>>(x, xln, ln2_w + l * Dc, ln2_b + l * Dc);

    // fc + bias + gelu
    gemm_wmma<5, false><<<dim3(tilesN256(D4), tilesM128(BS), 1), blk, 0, stream>>>(
        xln, Wf, h4, bf, BS, D4, Dc, Dc, D4, D4,
        0, 0, 0, 0, 0, 0, 1, 1.0f);

    // fcp + bias + residual into x
    gemm_wmma<3, false><<<dim3(tilesN256(Dc), tilesM128(BS), 1), blk, 0, stream>>>(
        h4, Wg, x, bg, BS, Dc, D4, D4, Dc, Dc,
        0, 0, 0, 0, 0, 0, 1, 1.0f);
  }

  // final LN
  layernorm_k<<<dim3(BS), blk, 0, stream>>>(x, xln, lnf_w, lnf_b);

  // gather rows at state positions (1::3)
  {
    int tot = Bc * Tc * Dc;
    gather_states_k<<<dim3((tot + 255) / 256), blk, 0, stream>>>(xln, xs);
  }

  // LM head: [512,1024] x lm_w^T -> [512, 50257]
  gemm_wmma<0, true><<<dim3(tilesN256(Vc), tilesM128(Bc * Tc), 1), blk, 0, stream>>>(
      xs, lm_w, out, nullptr, Bc * Tc, Vc, Dc, Dc, Dc, Vc,
      0, 0, 0, 0, 0, 0, 1, 1.0f);
}
